// SpatialGatingUnit_35304631173579
// MI455X (gfx1250) — 8-bit matrix operand order repaired and checked by reading, compiled, not run on hardware
//
#include <hip/hip_runtime.h>
#include <cstdint>
#include <cstddef>

#define DIM    2048
#define SEQ    2048
#define DHALF  1024
#define LN_EPS 1e-5f

#define W_SCALE   2097152.0f
#define W_INV     4.76837158203125e-7f

typedef __attribute__((ext_vector_type(16))) int   v16i;
typedef __attribute__((ext_vector_type(8)))  float v8f;

union Frag16 { v16i v; int2 d2[8]; int4 d4[4]; };

__device__ __forceinline__ unsigned int f2fp8_sw(float f) {
  union { float f; unsigned u; } v; v.f = f;
  const unsigned sgn = (v.u >> 24) & 0x80u;
  const unsigned a = v.u & 0x7FFFFFFFu;
  if (a >= 0x43E00000u) return sgn | 0x7Eu;
  const int E = (int)(a >> 23) - 127;
  const unsigned m = a & 0x7FFFFFu;
  if (E >= -6) {
    unsigned keep = m >> 20;
    const unsigned rest = m & 0xFFFFFu;
    keep += (rest > 0x80000u) || (rest == 0x80000u && (keep & 1u));
    unsigned code = ((unsigned)(E + 7) << 3) + keep;
    if (code > 0x7Eu) code = 0x7Eu;
    return sgn | code;
  }
  if (E >= -9) {
    const unsigned full = 0x800000u | m;
    const int sh = 20 + (-6 - E);
    unsigned keep = full >> sh;
    const unsigned rem = full & ((1u << sh) - 1u);
    const unsigned half = 1u << (sh - 1);
    keep += (rem > half) || (rem == half && (keep & 1u));
    return sgn | keep;
  }
  return sgn;
}

__device__ __forceinline__ unsigned pk4_fp8(float a, float b, float c, float d) {
#if __has_builtin(__builtin_amdgcn_cvt_pk_fp8_f32)
  int p = 0;
  p = __builtin_amdgcn_cvt_pk_fp8_f32(a, b, p, false);
  p = __builtin_amdgcn_cvt_pk_fp8_f32(c, d, p, true);
  return (unsigned)p;
#else
  return f2fp8_sw(a) | (f2fp8_sw(b) << 8) | (f2fp8_sw(c) << 16) | (f2fp8_sw(d) << 24);
#endif
}

__global__ __launch_bounds__(256)
void sgu_ln_transpose(const float* __restrict__ x,
                      const float* __restrict__ ln_scale,
                      const float* __restrict__ ln_bias,
                      unsigned char* __restrict__ gt8) {
  __shared__ unsigned char tile[32 * 1028];
  const int lane = threadIdx.x & 31;
  const int wave = threadIdx.x >> 5;
  const int t0   = blockIdx.x * 32;
  const int b    = blockIdx.y;

  #pragma unroll
  for (int r = 0; r < 4; ++r) {
    const int tt = wave * 4 + r;
    const float* row = x + ((size_t)(b * SEQ + t0 + tt)) * DIM + DHALF;

    float4 v[8];
    float s = 0.f, q = 0.f;
    #pragma unroll
    for (int i = 0; i < 8; ++i) {
      v[i] = *(const float4*)(row + lane * 4 + 128 * i);
      s += v[i].x + v[i].y + v[i].z + v[i].w;
      q += v[i].x * v[i].x + v[i].y * v[i].y + v[i].z * v[i].z + v[i].w * v[i].w;
    }
    #pragma unroll
    for (int off = 16; off > 0; off >>= 1) {
      s += __shfl_xor(s, off, 32);
      q += __shfl_xor(q, off, 32);
    }
    const float mu  = s * (1.0f / DHALF);
    const float var = q * (1.0f / DHALF) - mu * mu;
    const float inv = rsqrtf(var + LN_EPS);

    #pragma unroll
    for (int i = 0; i < 8; ++i) {
      const int d = lane * 4 + 128 * i;
      const float4 sc = *(const float4*)(ln_scale + d);
      const float4 bi = *(const float4*)(ln_bias + d);
      const unsigned p = pk4_fp8((v[i].x - mu) * inv * sc.x + bi.x,
                                 (v[i].y - mu) * inv * sc.y + bi.y,
                                 (v[i].z - mu) * inv * sc.z + bi.z,
                                 (v[i].w - mu) * inv * sc.w + bi.w);
      *(unsigned*)(tile + tt * 1028 + d) = p;
    }
  }
  __syncthreads();

  for (int d = wave; d < DHALF; d += 8) {
    gt8[((size_t)(b * DHALF + d)) * SEQ + t0 + lane] = tile[lane * 1028 + d];
  }
}

__global__ __launch_bounds__(256)
void sgu_wcast(const float* __restrict__ w, unsigned char* __restrict__ w8) {
  const size_t i = ((size_t)blockIdx.x * 256 + threadIdx.x) * 4;
  const float4 v = *(const float4*)(w + i);
  *(unsigned*)(w8 + i) = pk4_fp8(v.x * W_SCALE, v.y * W_SCALE,
                                 v.z * W_SCALE, v.w * W_SCALE);
}

__global__ __launch_bounds__(256)
void sgu_gemm(const unsigned char* __restrict__ w8,
              const unsigned char* __restrict__ gt8,
              const float* __restrict__ proj_b,
              const float* __restrict__ x,
              float* __restrict__ out) {
  const int lane = threadIdx.x & 31;
  const int wave = threadIdx.x >> 5;
  const int wm   = wave & 3;
  const int wn   = wave >> 2;
  const int b    = blockIdx.z;
  const int s0   = blockIdx.y * 128 + wm * 32;
  const int n0   = blockIdx.x * 128 + wn * 64;
  const int half = lane >> 4;
  const int l16  = lane & 15;

  const unsigned char* aRow = w8 + (size_t)(s0 + l16) * SEQ + half * 16;
  const unsigned char* bRow = gt8 + ((size_t)(b * DHALF + n0 + l16)) * SEQ + half * 16;

  v8f acc[2][4] = {};

  for (int k = 0; k < SEQ; k += 128) {
    Frag16 a[2], bb[4];
    #pragma unroll
    for (int j = 0; j < 8; ++j) {
      a[0].d2[j] = *(const int2*)(aRow + k + 32 * (j >> 1) + 8 * (j & 1));
      a[1].d2[j] = *(const int2*)(aRow + (size_t)16 * SEQ + k + 32 * (j >> 1) + 8 * (j & 1));
    }
    #pragma unroll
    for (int n = 0; n < 4; ++n) {
      #pragma unroll
      for (int j = 0; j < 4; ++j)
        bb[n].d4[j] = *(const int4*)(bRow + (size_t)n * 16 * SEQ + k + 32 * j);
    }
    #pragma unroll
    for (int n = 0; n < 4; ++n) {
      acc[0][n] = __builtin_amdgcn_wmma_f32_16x16x128_fp8_fp8(
          a[0].v, bb[n].v, (short)0, acc[0][n], false, false);
      acc[1][n] = __builtin_amdgcn_wmma_f32_16x16x128_fp8_fp8(
          a[1].v, bb[n].v, (short)0, acc[1][n], false, false);
    }
  }

  const float* xr = x   + (size_t)b * SEQ * DIM;
  float*       op = out + (size_t)b * SEQ * DHALF;
  #pragma unroll
  for (int m = 0; m < 2; ++m) {
    const int sBase = s0 + m * 16 + half * 8;
    #pragma unroll
    for (int n = 0; n < 4; ++n) {
      const int d = n0 + n * 16 + l16;
      #pragma unroll
      for (int j = 0; j < 8; ++j) {
        const int s = sBase + j;
        const float g = acc[m][n][j] * W_INV + proj_b[s];
        op[(size_t)s * DHALF + d] = g * xr[(size_t)s * DIM + d];
      }
    }
  }
}

extern "C" void kernel_launch(void* const* d_in, const int* in_sizes, int n_in,
                              void* d_out, int out_size, void* d_ws, size_t ws_size,
                              hipStream_t stream) {
  const float* x        = (const float*)d_in[0];
  const float* proj_w   = (const float*)d_in[1];
  const float* proj_b   = (const float*)d_in[2];
  const float* ln_scale = (const float*)d_in[3];
  const float* ln_bias  = (const float*)d_in[4];
  float* out = (float*)d_out;

  unsigned char* gt8 = (unsigned char*)d_ws;
  unsigned char* w8  = gt8 + (size_t)8 * DHALF * SEQ;

  sgu_ln_transpose<<<dim3(SEQ / 32, 8), 256, 0, stream>>>(x, ln_scale, ln_bias, gt8);
  sgu_wcast<<<(SEQ * SEQ) / 1024, 256, 0, stream>>>(proj_w, w8);
  sgu_gemm<<<dim3(DHALF / 128, SEQ / 128, 8), 256, 0, stream>>>(w8, gt8, proj_b, x, out);
}
